// Net_rumor_48086453846020
// MI455X (gfx1250) — compile-verified, matrix operand order checked by reading, not run on hardware
//
#include <hip/hip_runtime.h>
#include <hip/hip_bf16.h>

typedef __attribute__((ext_vector_type(16))) __bf16 v16bf;
typedef __attribute__((ext_vector_type(8)))  float  v8f;

#define N_NODES 50000
#define SEQ_LEN 32
#define EMBED   300
#define NHID    128
#define G4      512      // 4*NHID
#define NCLASS  4
#define N_EDGES 800000
#define MTILES  3125     // 50000 / 16
#define KXC     10       // ceil(300/32) K-chunks for x@Wih^T
#define KHC     4        // 128/32     K-chunks for h@Whh^T
#define NTG     32       // 512/16     column tiles of gate matrix
#define KC1     8        // 256/32     K-chunks GEMM1
#define NT1     8        // 128/16     col tiles GEMM1
#define KC2     4        // 128/32     K-chunks GEMM2

// ---------- helpers ----------
static __device__ __forceinline__ unsigned short f2bf(float f) {
  union { float f; unsigned u; } c; c.f = f;
  unsigned u = c.u + 0x7FFFu + ((c.u >> 16) & 1u);   // round-to-nearest-even
  return (unsigned short)(u >> 16);
}
// K-position of element e (0..15) within a 16-bit WMMA A/B fragment half
// (ISA 7.12.2: VGPR0..3 hold K 0..7 (+8 for lanes 16-31), VGPR4..7 hold K 16..23 (+8))
static __device__ __forceinline__ int klocal(int e, int hf) {
  return ((e & 8) << 1) + hf * 8 + (e & 7);
}
static __device__ __forceinline__ float sigf(float x) {
  return 1.0f / (1.0f + __expf(-x));
}
static __device__ __forceinline__ v8f wmma_bf16(v16bf a, v16bf b, v8f c) {
  return __builtin_amdgcn_wmma_f32_16x16x32_bf16(false, a, false, b, (short)0, c,
                                                 false, false);
}
static __device__ __forceinline__ void atomAddF(float* p, float v) {
  unsafeAtomicAdd(p, v);   // lowers to global_atomic_add_f32 on gfx1250
}

// ---------- generic zero ----------
__global__ void zero_f32(float* __restrict__ p, int n) {
  int i = blockIdx.x * blockDim.x + threadIdx.x;
  int stride = gridDim.x * blockDim.x;
  for (; i < n; i += stride) p[i] = 0.0f;
}

// ---------- weight repack into fragment-packed bf16 layout ----------
// BxPacked: [dir][nt=32][kc=10][lane=32][e=16]   B[k][n] = Wih[n][k], K padded 300->320
__global__ void repack_wih(const float* __restrict__ Wf, const float* __restrict__ Wb,
                           unsigned short* __restrict__ dst) {
  int idx = blockIdx.x * blockDim.x + threadIdx.x;
  const int per_dir = NTG * KXC * 512;
  if (idx >= 2 * per_dir) return;
  int dir = idx / per_dir;
  int rem = idx - dir * per_dir;
  int nt  = rem / (KXC * 512);
  int r2  = rem % (KXC * 512);
  int kc  = r2 >> 9;
  int ln  = (r2 >> 4) & 31;
  int e   = r2 & 15;
  int n   = nt * 16 + (ln & 15);
  int k   = kc * 32 + klocal(e, ln >> 4);
  const float* W = dir ? Wb : Wf;
  float v = (k < EMBED) ? W[(size_t)n * EMBED + k] : 0.0f;
  dst[idx] = f2bf(v);
}

// BhPacked: [dir][nt=32][kc=4][lane][e]   B[k][n] = Whh[n][k]
__global__ void repack_whh(const float* __restrict__ Wf, const float* __restrict__ Wb,
                           unsigned short* __restrict__ dst) {
  int idx = blockIdx.x * blockDim.x + threadIdx.x;
  const int per_dir = NTG * KHC * 512;
  if (idx >= 2 * per_dir) return;
  int dir = idx / per_dir;
  int rem = idx - dir * per_dir;
  int nt  = rem / (KHC * 512);
  int r2  = rem % (KHC * 512);
  int kc  = r2 >> 9;
  int ln  = (r2 >> 4) & 31;
  int e   = r2 & 15;
  int n   = nt * 16 + (ln & 15);
  int k   = kc * 32 + klocal(e, ln >> 4);
  const float* W = dir ? Wb : Wf;
  dst[idx] = f2bf(W[(size_t)n * NHID + k]);
}

// B1Packed: [nt=8][kc=8][lane][e]   W1 is [256][128] = B directly
__global__ void repack_w1(const float* __restrict__ W1, unsigned short* __restrict__ dst) {
  int idx = blockIdx.x * blockDim.x + threadIdx.x;
  if (idx >= NT1 * KC1 * 512) return;
  int nt = idx / (KC1 * 512);
  int r2 = idx % (KC1 * 512);
  int kc = r2 >> 9;
  int ln = (r2 >> 4) & 31;
  int e  = r2 & 15;
  int n  = nt * 16 + (ln & 15);
  int k  = kc * 32 + klocal(e, ln >> 4);
  dst[idx] = f2bf(W1[(size_t)k * NHID + n]);
}

// B2Packed: [kc=4][lane][e]   W2 is [128][4]; cols >=4 zero
__global__ void repack_w2(const float* __restrict__ W2, unsigned short* __restrict__ dst) {
  int idx = blockIdx.x * blockDim.x + threadIdx.x;
  if (idx >= KC2 * 512) return;
  int kc = idx >> 9;
  int ln = (idx >> 4) & 31;
  int e  = idx & 15;
  int n  = ln & 15;
  int k  = kc * 32 + klocal(e, ln >> 4);
  float v = (n < NCLASS) ? W2[(size_t)k * NCLASS + n] : 0.0f;
  dst[idx] = f2bf(v);
}

// ---------- persistent bidirectional LSTM over 16 nodes per workgroup ----------
// grid.x = MTILES, grid.y = 2 (direction), block = 256 (8 wave32)
__global__ __launch_bounds__(256) void lstm_kernel(
    const int* __restrict__ sentence, const float* __restrict__ embed,
    const unsigned short* __restrict__ BxPacked,
    const unsigned short* __restrict__ BhPacked,
    const float* __restrict__ b_f, const float* __restrict__ b_b,
    unsigned short* __restrict__ XcPacked) {
  __shared__ __align__(32) unsigned short aX[KXC * 512];   // x_t fragments (10 KB)
  __shared__ __align__(32) unsigned short aH[KHC * 512];   // h fragments   (4 KB)
  __shared__ float zbuf[16 * G4];                          // gate pre-acts (32 KB)
  __shared__ float cbuf[16 * NHID];                        // cell state    (8 KB)
  __shared__ int   toks[16];

  const int mtile = blockIdx.x;
  const int dir   = blockIdx.y;
  const int tid   = threadIdx.x;
  const int wave  = tid >> 5;
  const int lane  = tid & 31;
  const int half  = lane >> 4;
  const int l15   = lane & 15;

  const unsigned short* Bx = BxPacked + (size_t)dir * (NTG * KXC * 512);
  const unsigned short* Bh = BhPacked + (size_t)dir * (NTG * KHC * 512);
  const float* bias = dir ? b_b : b_f;

  for (int i = tid; i < KHC * 512; i += 256) aH[i] = 0;
  for (int i = tid; i < 16 * NHID; i += 256) cbuf[i] = 0.0f;
  __syncthreads();

  #pragma unroll 1
  for (int step = 0; step < SEQ_LEN; ++step) {
    const int t = dir ? (SEQ_LEN - 1 - step) : step;
    // stage token ids
    if (tid < 16) toks[tid] = sentence[(size_t)(mtile * 16 + tid) * SEQ_LEN + t];
    __syncthreads();
    // gather embedding rows into fragment-packed bf16 LDS
    for (int idx = tid; idx < KXC * 512; idx += 256) {
      int kc = idx >> 9;
      int r2 = idx & 511;
      int ln = r2 >> 4;
      int e  = r2 & 15;
      int m  = ln & 15;
      int k  = kc * 32 + klocal(e, ln >> 4);
      float v = 0.0f;
      if (k < EMBED) v = embed[(size_t)toks[m] * EMBED + k];
      aX[idx] = f2bf(v);
    }
    __syncthreads();

    // z = x@Wih^T + h@Whh^T + b : each wave owns 4 of 32 column tiles.
    // NOTE: tt is kept as a real loop (unroll 1) so the compiler cannot hoist
    // all 56 B-weight fragments across the timestep loop (which previously
    // caused >256-VGPR extended addressing + scratch spills in the hot loop).
    #pragma unroll 1
    for (int tt = 0; tt < 4; ++tt) {
      const int nt = wave * 4 + tt;
      const float bcol = bias[nt * 16 + l15];
      v8f acc = {bcol, bcol, bcol, bcol, bcol, bcol, bcol, bcol};
      const unsigned short* bxp = Bx + (size_t)nt * KXC * 512;
      for (int kc = 0; kc < KXC; ++kc) {
        if (kc + 1 < KXC)
          __builtin_prefetch(bxp + (kc + 1) * 512 + lane * 16, 0, 1);
        v16bf a = *(const v16bf*)(aX + kc * 512 + lane * 16);
        v16bf b = *(const v16bf*)(bxp + kc * 512 + lane * 16);
        acc = wmma_bf16(a, b, acc);
      }
      const unsigned short* bhp = Bh + (size_t)nt * KHC * 512;
      for (int kc = 0; kc < KHC; ++kc) {
        v16bf a = *(const v16bf*)(aH + kc * 512 + lane * 16);
        v16bf b = *(const v16bf*)(bhp + kc * 512 + lane * 16);
        acc = wmma_bf16(a, b, acc);
      }
      #pragma unroll
      for (int r = 0; r < 8; ++r)
        zbuf[(r + half * 8) * G4 + nt * 16 + l15] = acc[r];
    }
    __syncthreads();

    // gate math (torch order i,f,g,o); write h back in A-fragment layout
    for (int i = tid; i < 16 * NHID; i += 256) {
      int m = i >> 7;
      int j = i & 127;
      float zi = zbuf[m * G4 + j];
      float zf = zbuf[m * G4 + NHID + j];
      float zg = zbuf[m * G4 + 2 * NHID + j];
      float zo = zbuf[m * G4 + 3 * NHID + j];
      float c  = sigf(zf) * cbuf[i] + sigf(zi) * tanhf(zg);
      float h  = sigf(zo) * tanhf(c);
      cbuf[i] = c;
      int kc = j >> 5;
      int kl = j & 31;
      int hf = (kl >> 3) & 1;
      int e  = ((kl & 16) >> 1) | (kl & 7);
      aH[kc * 512 + (hf * 16 + m) * 16 + e] = f2bf(h);
    }
    __syncthreads();
  }

  // final hidden -> XcPacked: fwd occupies k-chunks 0..3, bwd 4..7
  for (int i = tid; i < KHC * 512; i += 256) {
    int kc = i >> 9;
    int r2 = i & 511;
    XcPacked[((size_t)mtile * (2 * KHC) + dir * KHC + kc) * 512 + r2] = aH[i];
  }
}

// ---------- GEMM1: xw1[N,128] = Xc[N,256] @ W1 ----------
__global__ __launch_bounds__(256) void gemm1_kernel(
    const unsigned short* __restrict__ XcPacked,
    const unsigned short* __restrict__ B1Packed,
    float* __restrict__ xw1) {
  const int mtile = blockIdx.x;
  const int wave  = threadIdx.x >> 5;      // one col tile per wave (8 tiles)
  const int lane  = threadIdx.x & 31;
  const int half  = lane >> 4;
  const int l15   = lane & 15;
  v8f acc = {0.f, 0.f, 0.f, 0.f, 0.f, 0.f, 0.f, 0.f};
  for (int kc = 0; kc < KC1; ++kc) {
    v16bf a = *(const v16bf*)(XcPacked + ((size_t)mtile * KC1 + kc) * 512 + lane * 16);
    v16bf b = *(const v16bf*)(B1Packed + ((size_t)wave * KC1 + kc) * 512 + lane * 16);
    acc = wmma_bf16(a, b, acc);
  }
  #pragma unroll
  for (int r = 0; r < 8; ++r)
    xw1[(size_t)(mtile * 16 + r + half * 8) * NHID + wave * 16 + l15] = acc[r];
}

// ---------- scatter-add layer 1: one wave per edge, 4 floats per lane ----------
__global__ __launch_bounds__(256) void scatter1_kernel(
    const int* __restrict__ ei, const float* __restrict__ xw1,
    float* __restrict__ acc1) {
  const int e = blockIdx.x * 8 + (threadIdx.x >> 5);
  if (e >= N_EDGES) return;
  const int lane = threadIdx.x & 31;
  const int src = ei[e];
  const int dst = ei[N_EDGES + e];
  const float* s = xw1 + (size_t)src * NHID + lane * 4;
  float* d = acc1 + (size_t)dst * NHID + lane * 4;
  #pragma unroll
  for (int c = 0; c < 4; ++c) atomAddF(d + c, s[c]);
}

// ---------- ReLU + pack into A fragments for GEMM2 ----------
__global__ void relu_pack_kernel(const float* __restrict__ acc1,
                                 unsigned short* __restrict__ xr) {
  int idx = blockIdx.x * blockDim.x + threadIdx.x;
  int stride = gridDim.x * blockDim.x;
  const int total = MTILES * KC2 * 512;
  for (; idx < total; idx += stride) {
    int mtile = idx / (KC2 * 512);
    int r2    = idx % (KC2 * 512);
    int kc = r2 >> 9;
    int ln = (r2 >> 4) & 31;
    int e  = r2 & 15;
    int m  = ln & 15;
    int k  = kc * 32 + klocal(e, ln >> 4);
    float v = acc1[(size_t)(mtile * 16 + m) * NHID + k];
    xr[idx] = f2bf(v > 0.0f ? v : 0.0f);
  }
}

// ---------- GEMM2: xw2[N,4] = relu(acc1) @ W2 (one wave per row tile) ----------
__global__ __launch_bounds__(256) void gemm2_kernel(
    const unsigned short* __restrict__ xr,
    const unsigned short* __restrict__ B2Packed,
    float* __restrict__ xw2) {
  const int mtile = blockIdx.x * 8 + (threadIdx.x >> 5);
  if (mtile >= MTILES) return;
  const int lane = threadIdx.x & 31;
  const int half = lane >> 4;
  const int l15  = lane & 15;
  v8f acc = {0.f, 0.f, 0.f, 0.f, 0.f, 0.f, 0.f, 0.f};
  for (int kc = 0; kc < KC2; ++kc) {
    v16bf a = *(const v16bf*)(xr + ((size_t)mtile * KC2 + kc) * 512 + lane * 16);
    v16bf b = *(const v16bf*)(B2Packed + (size_t)kc * 512 + lane * 16);
    acc = wmma_bf16(a, b, acc);
  }
  if (l15 < NCLASS) {
    #pragma unroll
    for (int r = 0; r < 8; ++r)
      xw2[(size_t)(mtile * 16 + r + half * 8) * NCLASS + l15] = acc[r];
  }
}

// ---------- scatter-add layer 2: one thread per edge ----------
__global__ void scatter2_kernel(const int* __restrict__ ei,
                                const float* __restrict__ xw2,
                                float* __restrict__ out) {
  int e = blockIdx.x * blockDim.x + threadIdx.x;
  if (e >= N_EDGES) return;
  const int src = ei[e];
  const int dst = ei[N_EDGES + e];
  #pragma unroll
  for (int c = 0; c < NCLASS; ++c)
    atomAddF(out + (size_t)dst * NCLASS + c, xw2[(size_t)src * NCLASS + c]);
}

// ---------- host ----------
extern "C" void kernel_launch(void* const* d_in, const int* in_sizes, int n_in,
                              void* d_out, int out_size, void* d_ws, size_t ws_size,
                              hipStream_t stream) {
  const int*   sentence = (const int*)d_in[0];
  const int*   ei1      = (const int*)d_in[1];
  const int*   ei2      = (const int*)d_in[2];
  // d_in[3], d_in[4]: edge weights (unused by reference GCN)
  const float* embed    = (const float*)d_in[5];
  const float* Wih_f    = (const float*)d_in[6];
  const float* Whh_f    = (const float*)d_in[7];
  const float* b_f      = (const float*)d_in[8];
  const float* Wih_b    = (const float*)d_in[9];
  const float* Whh_b    = (const float*)d_in[10];
  const float* b_b      = (const float*)d_in[11];
  const float* W1       = (const float*)d_in[12];
  const float* W2       = (const float*)d_in[13];
  float* out = (float*)d_out;

  char* ws = (char*)d_ws;
  unsigned short* BxP = (unsigned short*)(ws + 0);          //  655360 B
  unsigned short* BhP = (unsigned short*)(ws + 655360);     //  262144 B
  unsigned short* B1P = (unsigned short*)(ws + 917504);     //   65536 B
  unsigned short* B2P = (unsigned short*)(ws + 983040);     //    4096 B
  unsigned short* XcP = (unsigned short*)(ws + 987136);     // 25.6 MB
  float*          xw1 = (float*)(ws + 26587136);            // 25.6 MB
  float*          ac1 = (float*)(ws + 52187136);            // 25.6 MB
  unsigned short* xrP = (unsigned short*)(ws + 77787136);   // 12.8 MB
  float*          xw2 = (float*)(ws + 90587136);            //  0.8 MB

  // zero accumulators / output
  zero_f32<<<2048, 256, 0, stream>>>(ac1, N_NODES * NHID);
  zero_f32<<<512, 256, 0, stream>>>(out, N_NODES * NCLASS);

  // weight repack to WMMA fragment layout (bf16)
  repack_wih<<<(2 * NTG * KXC * 512 + 255) / 256, 256, 0, stream>>>(Wih_f, Wih_b, BxP);
  repack_whh<<<(2 * NTG * KHC * 512 + 255) / 256, 256, 0, stream>>>(Whh_f, Whh_b, BhP);
  repack_w1<<<(NT1 * KC1 * 512 + 255) / 256, 256, 0, stream>>>(W1, B1P);
  repack_w2<<<(KC2 * 512 + 255) / 256, 256, 0, stream>>>(W2, B2P);

  // bidirectional LSTM (16 nodes / workgroup, 8 waves, WMMA gate GEMMs)
  lstm_kernel<<<dim3(MTILES, 2), 256, 0, stream>>>(sentence, embed, BxP, BhP,
                                                   b_f, b_b, XcP);
  // GCN layer 1
  gemm1_kernel<<<MTILES, 256, 0, stream>>>(XcP, B1P, xw1);
  scatter1_kernel<<<N_EDGES / 8, 256, 0, stream>>>(ei1, xw1, ac1);
  relu_pack_kernel<<<4096, 256, 0, stream>>>(ac1, xrP);
  // GCN layer 2
  gemm2_kernel<<<(MTILES + 7) / 8, 256, 0, stream>>>(xrP, B2P, xw2);
  scatter2_kernel<<<(N_EDGES + 255) / 256, 256, 0, stream>>>(ei2, xw2, out);
  (void)in_sizes; (void)n_in; (void)out_size; (void)ws_size;
}
